// Attention_Head_42339787604540
// MI455X (gfx1250) — matrix operand order repaired and checked by reading, not compiled, not run on hardware
//
#include <hip/hip_runtime.h>

// ---------------------------------------------------------------------------
// Attention head for MI455X (gfx1250, wave32, WMMA, async global->LDS DMA).
// B=2, S=2048, E=2048, H=16, D=128. fp32 I/O, bf16 WMMA compute, f32 accum.
// ---------------------------------------------------------------------------

#define B_CONST 2
#define S_CONST 2048
#define E_CONST 2048
#define H_CONST 16
#define D_CONST 128
#define F_CONST (H_CONST * D_CONST)   // 2048
#define M_CONST (B_CONST * S_CONST)   // 4096

typedef __attribute__((ext_vector_type(16))) __bf16 v16bf;
typedef __attribute__((ext_vector_type(8)))  float  v8f;

union Frag16 {
  v16bf v;
  unsigned short u[16];
  uint4 q[2];
};

__device__ __forceinline__ unsigned short f32_to_bf16(float f) {
  unsigned int u = __float_as_uint(f);
  unsigned int r = 0x7FFFu + ((u >> 16) & 1u);   // round-to-nearest-even
  return (unsigned short)((u + r) >> 16);
}
__device__ __forceinline__ float bf16_to_f32(unsigned short h) {
  return __uint_as_float(((unsigned int)h) << 16);
}
__device__ __forceinline__ v8f zero8() {
  v8f z;
#pragma unroll
  for (int i = 0; i < 8; ++i) z[i] = 0.0f;
  return z;
}
__device__ __forceinline__ void load_frag(Frag16& f, const unsigned short* p0,
                                          const unsigned short* p1) {
  f.q[0] = *(const uint4*)(p0);
  f.q[1] = *(const uint4*)(p1);
}

// Async DMA: 16B per lane, global -> LDS, tracked with ASYNCcnt.
// VDST VGPR carries the LDS byte address (low 32 bits of the flat shared
// address == LDS offset, since the LDS aperture occupies addr[63:32]).
__device__ __forceinline__ void async_b128(const unsigned short* lds_ptr,
                                           const unsigned short* gptr) {
  unsigned loff = (unsigned)(uintptr_t)lds_ptr;
  asm volatile("global_load_async_to_lds_b128 %0, %1, off"
               :: "v"(loff), "v"(gptr)
               : "memory");
}

// ---------------------------------------------------------------------------
// 1) fp32 -> bf16 conversion
// ---------------------------------------------------------------------------
__global__ void cvt_f32_bf16(const float* __restrict__ in,
                             unsigned short* __restrict__ out, int n) {
  int i = blockIdx.x * blockDim.x + threadIdx.x;
  if (i < n) out[i] = f32_to_bf16(in[i]);
}

// ---------------------------------------------------------------------------
// 2) GEMM: Y[M,N] = A[M,K] * W[N,K]^T + bias[N]   (bf16 in, f32 acc, bf16 out)
//    Block = 8 waves -> 128x64 tile. Per K-step (32): async-DMA the next
//    A(128x32)/B(64x32) panels into LDS buffer buf^1 while WMMAs consume buf.
//    Fragment loads are staggered one WMMA ahead so each s_wait_dscnt is for
//    a load issued at least one matrix-op earlier (LDS latency hidden).
// ---------------------------------------------------------------------------
__global__ __launch_bounds__(256)
void gemm_bias_wmma(const unsigned short* __restrict__ A,
                    const unsigned short* __restrict__ W,
                    const float* __restrict__ bias,
                    unsigned short* __restrict__ Y,
                    int M, int N, int K) {
  __shared__ __align__(16) unsigned short Asb[2][128 * 32];   // 2 x 8 KB
  __shared__ __align__(16) unsigned short Bsb[2][64 * 32];    // 2 x 4 KB

  const int tid = threadIdx.x;
  const int lane = tid & 31;
  const int wave = tid >> 5;
  const int tiles_n = N >> 6;                     // N / 64
  const int tm = blockIdx.x / tiles_n;            // 128-row block tile
  const int tn = blockIdx.x - tm * tiles_n;
  const int m0 = tm << 7, n0 = tn << 6;

  // staging map: 256 threads -> 64 rows x 4 chunks of 8 halfs (16B)
  const int sr = tid >> 2;                        // 0..63
  const int sc = (tid & 3) << 3;                  // half offset 0,8,16,24

  const int l15 = lane & 15;
  const int kh8 = (lane >> 4) << 3;               // 0 or 8
  const int kh16 = (lane >> 4) << 4;              // 0 or 16

  v8f acc0 = zero8(), acc1 = zero8(), acc2 = zero8(), acc3 = zero8();

  // prologue: stage k=0 into buffer 0
  async_b128(&Asb[0][sr * 32 + sc],        A + (size_t)(m0 + sr) * K + sc);
  async_b128(&Asb[0][(64 + sr) * 32 + sc], A + (size_t)(m0 + 64 + sr) * K + sc);
  async_b128(&Bsb[0][sr * 32 + sc],        W + (size_t)(n0 + sr) * K + sc);

  int buf = 0;
  for (int k0 = 0; k0 < K; k0 += 32, buf ^= 1) {
    if (k0 + 32 < K) {
      const int kn = k0 + 32;
      async_b128(&Asb[buf ^ 1][sr * 32 + sc],
                 A + (size_t)(m0 + sr) * K + kn + sc);
      async_b128(&Asb[buf ^ 1][(64 + sr) * 32 + sc],
                 A + (size_t)(m0 + 64 + sr) * K + kn + sc);
      async_b128(&Bsb[buf ^ 1][sr * 32 + sc],
                 W + (size_t)(n0 + sr) * K + kn + sc);
      // async ops retire in order: <=3 outstanding => current buffer landed
      asm volatile("s_wait_asynccnt 0x3" ::: "memory");
    } else {
      asm volatile("s_wait_asynccnt 0x0" ::: "memory");
    }
    __syncthreads();                              // all waves' DMA visible

    const unsigned short* ar = &Asb[buf][(wave * 16 + l15) * 32];
    const unsigned short* br0 = &Bsb[buf][(0 * 16 + l15) * 32 + kh16];
    const unsigned short* br1 = &Bsb[buf][(1 * 16 + l15) * 32 + kh16];
    const unsigned short* br2 = &Bsb[buf][(2 * 16 + l15) * 32 + kh16];
    const unsigned short* br3 = &Bsb[buf][(3 * 16 + l15) * 32 + kh16];

    Frag16 af, b0, b1, b2, b3;
    load_frag(af, ar + kh16, ar + kh16 + 8);
    load_frag(b0, br0, br0 + 8);
    load_frag(b1, br1, br1 + 8);
    acc0 = __builtin_amdgcn_wmma_f32_16x16x32_bf16(
        false, af.v, false, b0.v, (short)0, acc0, false, false);
    load_frag(b2, br2, br2 + 8);
    acc1 = __builtin_amdgcn_wmma_f32_16x16x32_bf16(
        false, af.v, false, b1.v, (short)0, acc1, false, false);
    load_frag(b3, br3, br3 + 8);
    acc2 = __builtin_amdgcn_wmma_f32_16x16x32_bf16(
        false, af.v, false, b2.v, (short)0, acc2, false, false);
    acc3 = __builtin_amdgcn_wmma_f32_16x16x32_bf16(
        false, af.v, false, b3.v, (short)0, acc3, false, false);

    __syncthreads();  // reads of buf done before next iter's DMA overwrites it
  }

  const int m0w = m0 + wave * 16;
  v8f accs[4] = {acc0, acc1, acc2, acc3};
#pragma unroll
  for (int t = 0; t < 4; ++t) {
    const int n = n0 + t * 16 + l15;
    const float bv = bias[n];
#pragma unroll
    for (int r = 0; r < 8; ++r) {
      const int m = m0w + r + ((lane >> 4) << 3); // C layout: M=r / r+8
      Y[(size_t)m * N + n] = f32_to_bf16(accs[t][r] + bv);
    }
  }
}

// ---------------------------------------------------------------------------
// 3) RoPE (interleaved pairs) in place on bf16 [rows, 2048]
// ---------------------------------------------------------------------------
__global__ void rope_bf16(unsigned short* __restrict__ X, int rows) {
  const int p = blockIdx.x * blockDim.x + threadIdx.x;
  const int pairs_per_row = F_CONST / 2;          // 1024
  if (p >= rows * pairs_per_row) return;
  const int row = p / pairs_per_row;
  const int j = p - row * pairs_per_row;
  const int h = j >> 6;                           // 64 pairs per head
  const int i = j & 63;
  const int s = row & (S_CONST - 1);              // row = b*S + s
  // inv_freq = 10000^(-(2i)/128)
  const float inv = __expf(-((float)(2 * i) * (1.0f / 128.0f)) * 9.210340371976184f);
  const float ang = (float)s * inv;
  float sn, cs;
  __sincosf(ang, &sn, &cs);
  const size_t idx = (size_t)row * F_CONST + h * D_CONST + 2 * i;
  const float x0 = bf16_to_f32(X[idx]);
  const float x1 = bf16_to_f32(X[idx + 1]);
  X[idx]     = f32_to_bf16(x0 * cs - x1 * sn);
  X[idx + 1] = f32_to_bf16(x1 * cs + x0 * sn);
}

// ---------------------------------------------------------------------------
// 4) Causal flash attention. 4 waves/block; wave w owns q-tile qgroup*4+w.
//    K tile async-DMA'd row-major [32 x 128]; V tile staged TRANSPOSED
//    [128 x 32] so the P*V B-fragment is a contiguous ds_load. P is re-laid
//    out C-frag -> A-frag through per-wave LDS (in-order, s_wait_dscnt).
//    Fragment loads staggered one WMMA ahead in every chain.
// ---------------------------------------------------------------------------
__global__ __launch_bounds__(128)
void flash_attn_wmma(const unsigned short* __restrict__ Qb,
                     const unsigned short* __restrict__ Kb,
                     const unsigned short* __restrict__ Vb,
                     float* __restrict__ Out) {
  __shared__ __align__(16) unsigned short Klds[32 * 128];   // 8 KB
  __shared__ __align__(16) unsigned short Vt[128 * 32];     // 8 KB (transposed)
  __shared__ __align__(16) unsigned short Pt[4][16 * 32];   // 4 KB (per wave)

  const int lane = threadIdx.x & 31;
  const int wave = threadIdx.x >> 5;
  const int qgroup = blockIdx.x & 31;             // S/64 = 32 groups
  const int bh = blockIdx.x >> 5;
  const int b = bh >> 4, h = bh & 15;

  const int l15 = lane & 15;
  const int hi = lane >> 4;                       // 0/1 half-wave
  const int kh8 = hi << 3;

  const int qt = qgroup * 4 + wave;
  const int q0 = qt << 4;
  const size_t rowbase = (size_t)(b * S_CONST) * F_CONST + h * D_CONST;

  // --- load Q A-fragments (16 x 128 -> 4 x (16x32)) ---
  Frag16 qa[4];
  {
    const unsigned short* qp = Qb + rowbase + (size_t)(q0 + l15) * F_CONST;
#pragma unroll
    for (int c = 0; c < 4; ++c)
      load_frag(qa[c], qp + c * 32 + (hi << 4), qp + c * 32 + (hi << 4) + 8);
  }

  float mi[8], li[8], corr[8];
  v8f accO[8];
#pragma unroll
  for (int r = 0; r < 8; ++r) { mi[r] = -3.0e38f; li[r] = 0.0f; }
#pragma unroll
  for (int dc = 0; dc < 8; ++dc) accO[dc] = zero8();

  const int steps = qgroup * 2 + 2;               // keys 0 .. qgroup*64+63
  const int my_kmax = q0 + 15;
  const float scale = 0.08838834764831845f;       // 1/sqrt(128)

  for (int st = 0; st < steps; ++st) {
    const int k0 = st * 32;
    __syncthreads();                              // prev-step readers done
    // ---- stage K (async DMA, row major) and V (register transpose) ----
    {
      const int r = threadIdx.x >> 2;             // key row 0..31
      const int cc = (threadIdx.x & 3) << 5;      // half chunk 0,32,64,96
      const unsigned short* kp =
          Kb + rowbase + (size_t)(k0 + r) * F_CONST + cc;
      const unsigned short* vp =
          Vb + rowbase + (size_t)(k0 + r) * F_CONST + cc;
      if (st + 1 < steps) {                       // hint next tile
        __builtin_prefetch(kp + 32 * F_CONST, 0, 3);
        __builtin_prefetch(vp + 32 * F_CONST, 0, 3);
      }
      const unsigned short* kl = &Klds[r * 128 + cc];
      async_b128(kl,      kp);
      async_b128(kl + 8,  kp + 8);
      async_b128(kl + 16, kp + 16);
      async_b128(kl + 24, kp + 24);
      union { uint4 q[4]; unsigned short u[32]; } vt;
      vt.q[0] = *(const uint4*)(vp);
      vt.q[1] = *(const uint4*)(vp + 8);
      vt.q[2] = *(const uint4*)(vp + 16);
      vt.q[3] = *(const uint4*)(vp + 24);
#pragma unroll
      for (int jj = 0; jj < 32; ++jj)
        Vt[(cc + jj) * 32 + r] = vt.u[jj];
    }
    asm volatile("s_wait_asynccnt 0x0" ::: "memory");
    __syncthreads();
    if (k0 > my_kmax) continue;                   // wave-uniform skip

    // ---- scores S = Q K^T (two 16x16 C-frags, keys g*16..) ----
    v8f sc[2];
#pragma unroll
    for (int g = 0; g < 2; ++g) {
      const unsigned short* kr = &Klds[(g * 16 + l15) * 128 + (hi << 4)];
      Frag16 f0, f1, f2, f3;
      v8f s = zero8();
      load_frag(f0, kr,       kr + 8);
      load_frag(f1, kr + 32,  kr + 40);
      s = __builtin_amdgcn_wmma_f32_16x16x32_bf16(
          false, qa[0].v, false, f0.v, (short)0, s, false, false);
      load_frag(f2, kr + 64,  kr + 72);
      s = __builtin_amdgcn_wmma_f32_16x16x32_bf16(
          false, qa[1].v, false, f1.v, (short)0, s, false, false);
      load_frag(f3, kr + 96,  kr + 104);
      s = __builtin_amdgcn_wmma_f32_16x16x32_bf16(
          false, qa[2].v, false, f2.v, (short)0, s, false, false);
      s = __builtin_amdgcn_wmma_f32_16x16x32_bf16(
          false, qa[3].v, false, f3.v, (short)0, s, false, false);
      sc[g] = s;
    }

    // ---- scale + causal mask ----
#pragma unroll
    for (int g = 0; g < 2; ++g) {
#pragma unroll
      for (int r = 0; r < 8; ++r) {
        const int key = k0 + g * 16 + l15;
        const int q = q0 + r + (hi << 3);
        float sv = sc[g][r] * scale;
        if (key > q) sv = -3.0e38f;
        sc[g][r] = sv;
      }
    }

    // ---- online softmax: row reductions across 16-lane half-groups ----
#pragma unroll
    for (int r = 0; r < 8; ++r) {
      float rm = fmaxf(sc[0][r], sc[1][r]);
      rm = fmaxf(rm, __shfl_xor(rm, 1, 32));
      rm = fmaxf(rm, __shfl_xor(rm, 2, 32));
      rm = fmaxf(rm, __shfl_xor(rm, 4, 32));
      rm = fmaxf(rm, __shfl_xor(rm, 8, 32));
      const float mnew = fmaxf(mi[r], rm);
      corr[r] = __expf(mi[r] - mnew);
      const float p0 = __expf(sc[0][r] - mnew);
      const float p1 = __expf(sc[1][r] - mnew);
      float rs = p0 + p1;
      rs += __shfl_xor(rs, 1, 32);
      rs += __shfl_xor(rs, 2, 32);
      rs += __shfl_xor(rs, 4, 32);
      rs += __shfl_xor(rs, 8, 32);
      li[r] = li[r] * corr[r] + rs;
      mi[r] = mnew;
      sc[0][r] = p0;
      sc[1][r] = p1;
    }
#pragma unroll
    for (int dc = 0; dc < 8; ++dc)
#pragma unroll
      for (int r = 0; r < 8; ++r) accO[dc][r] *= corr[r];

    // ---- P: C-layout -> A-layout via per-wave LDS ----
#pragma unroll
    for (int g = 0; g < 2; ++g)
#pragma unroll
      for (int r = 0; r < 8; ++r)
        Pt[wave][(r + (hi << 3)) * 32 + g * 16 + l15] = f32_to_bf16(sc[g][r]);
    asm volatile("s_wait_dscnt 0" ::: "memory");  // per-wave LDS RAW

    Frag16 pa;
    {
      const unsigned short* pp = &Pt[wave][l15 * 32];
      load_frag(pa, pp + (hi << 4), pp + (hi << 4) + 8);
    }

    // ---- O += P * V  (staggered V-frag loads, 8-WMMA chain) ----
    {
      const unsigned short* vb = &Vt[l15 * 32 + (hi << 4)];
      Frag16 v0, v1, v2, v3;
      load_frag(v0, vb,            vb + 8);            // dc=0 (+512/dc)
      load_frag(v1, vb + 1 * 512,  vb + 1 * 512 + 8);
      accO[0] = __builtin_amdgcn_wmma_f32_16x16x32_bf16(
          false, pa.v, false, v0.v, (short)0, accO[0], false, false);
      load_frag(v2, vb + 2 * 512,  vb + 2 * 512 + 8);
      accO[1] = __builtin_amdgcn_wmma_f32_16x16x32_bf16(
          false, pa.v, false, v1.v, (short)0, accO[1], false, false);
      load_frag(v3, vb + 3 * 512,  vb + 3 * 512 + 8);
      accO[2] = __builtin_amdgcn_wmma_f32_16x16x32_bf16(
          false, pa.v, false, v2.v, (short)0, accO[2], false, false);
      load_frag(v0, vb + 4 * 512,  vb + 4 * 512 + 8);
      accO[3] = __builtin_amdgcn_wmma_f32_16x16x32_bf16(
          false, pa.v, false, v3.v, (short)0, accO[3], false, false);
      load_frag(v1, vb + 5 * 512,  vb + 5 * 512 + 8);
      accO[4] = __builtin_amdgcn_wmma_f32_16x16x32_bf16(
          false, pa.v, false, v0.v, (short)0, accO[4], false, false);
      load_frag(v2, vb + 6 * 512,  vb + 6 * 512 + 8);
      accO[5] = __builtin_amdgcn_wmma_f32_16x16x32_bf16(
          false, pa.v, false, v1.v, (short)0, accO[5], false, false);
      load_frag(v3, vb + 7 * 512,  vb + 7 * 512 + 8);
      accO[6] = __builtin_amdgcn_wmma_f32_16x16x32_bf16(
          false, pa.v, false, v2.v, (short)0, accO[6], false, false);
      accO[7] = __builtin_amdgcn_wmma_f32_16x16x32_bf16(
          false, pa.v, false, v3.v, (short)0, accO[7], false, false);
    }
  }

  // ---- normalize and store fp32 output [b, s, h*128 + d] ----
#pragma unroll
  for (int dc = 0; dc < 8; ++dc) {
#pragma unroll
    for (int r = 0; r < 8; ++r) {
      const int m = r + (hi << 3);
      Out[(size_t)(b * S_CONST + q0 + m) * F_CONST + h * D_CONST + dc * 16 + l15] =
          accO[dc][r] / li[r];
    }
  }
}

// ---------------------------------------------------------------------------
// Host-side orchestration (graph-capture safe: only kernel launches on stream)
// ---------------------------------------------------------------------------
extern "C" void kernel_launch(void* const* d_in, const int* in_sizes, int n_in,
                              void* d_out, int out_size, void* d_ws, size_t ws_size,
                              hipStream_t stream) {
  const float* logits = (const float*)d_in[0];
  const float* Wq = (const float*)d_in[1];
  const float* bq = (const float*)d_in[2];
  const float* Wk = (const float*)d_in[3];
  const float* bk = (const float*)d_in[4];
  const float* Wv = (const float*)d_in[5];
  const float* bv = (const float*)d_in[6];
  float* out = (float*)d_out;

  // workspace layout (bf16 halves), ~88 MB total
  unsigned short* Xb  = (unsigned short*)d_ws;                  // [4096,2048]
  unsigned short* Wqb = Xb  + (size_t)M_CONST * E_CONST;
  unsigned short* Wkb = Wqb + (size_t)F_CONST * E_CONST;
  unsigned short* Wvb = Wkb + (size_t)F_CONST * E_CONST;
  unsigned short* Qb  = Wvb + (size_t)F_CONST * E_CONST;        // [4096,2048]
  unsigned short* Kb  = Qb  + (size_t)M_CONST * F_CONST;
  unsigned short* Vb  = Kb  + (size_t)M_CONST * F_CONST;

  const int nX = M_CONST * E_CONST;    // 8.39M
  const int nW = F_CONST * E_CONST;    // 4.19M
  cvt_f32_bf16<<<(nX + 255) / 256, 256, 0, stream>>>(logits, Xb, nX);
  cvt_f32_bf16<<<(nW + 255) / 256, 256, 0, stream>>>(Wq, Wqb, nW);
  cvt_f32_bf16<<<(nW + 255) / 256, 256, 0, stream>>>(Wk, Wkb, nW);
  cvt_f32_bf16<<<(nW + 255) / 256, 256, 0, stream>>>(Wv, Wvb, nW);

  // GEMMs: (M/128)*(N/64) = 1024 blocks of 8 waves (128x64 tile per block)
  const int gemm_blocks = (M_CONST / 128) * (F_CONST / 64);
  gemm_bias_wmma<<<gemm_blocks, 256, 0, stream>>>(Xb, Wqb, bq, Qb,
                                                  M_CONST, F_CONST, E_CONST);
  gemm_bias_wmma<<<gemm_blocks, 256, 0, stream>>>(Xb, Wkb, bk, Kb,
                                                  M_CONST, F_CONST, E_CONST);
  gemm_bias_wmma<<<gemm_blocks, 256, 0, stream>>>(Xb, Wvb, bv, Vb,
                                                  M_CONST, F_CONST, E_CONST);

  // RoPE on Q and K
  const int pairs = M_CONST * (F_CONST / 2);
  rope_bf16<<<(pairs + 255) / 256, 256, 0, stream>>>(Qb, M_CONST);
  rope_bf16<<<(pairs + 255) / 256, 256, 0, stream>>>(Kb, M_CONST);

  // Flash attention: B*H * (S/64) = 1024 blocks of 4 waves
  flash_attn_wmma<<<B_CONST * H_CONST * (S_CONST / 64), 128, 0, stream>>>(
      Qb, Kb, Vb, out);
}
